// MultiHeadAttention_61813169324592
// MI455X (gfx1250) — matrix operand order repaired and checked by reading, not compiled, not run on hardware
//
#include <hip/hip_runtime.h>
#include <hip/hip_bf16.h>
#include <math.h>

// ---------------------------------------------------------------------------
// MHA forward for MI455X (gfx1250, wave32, WMMA bf16 with fp32 accumulate)
// ---------------------------------------------------------------------------

typedef __bf16 bf16;
typedef bf16  v16bf  __attribute__((ext_vector_type(16)));
typedef bf16  bf16x2 __attribute__((ext_vector_type(2)));
typedef bf16  bf16x8 __attribute__((ext_vector_type(8)));
typedef float v8f    __attribute__((ext_vector_type(8)));

#define B_SZ   2
#define SEQ    2048
#define NH     16
#define HD     64
#define DM     1024
#define M_ROWS (B_SZ * SEQ)   // 4096

// 16-bit A-matrix 16x32 fragment K mapping (ISA 7.12.2):
// lanes 0-15: V0..3 -> K=0..7, V4..7 -> K=16..23 ; lanes 16-31: +8
static __device__ __forceinline__ int a_kmap(int v, int half) {
    return 2 * (v & 3) + ((v & 4) ? 16 : 0) + 8 * half;
}
// 16-bit B-matrix 32x16 fragment K mapping (same as A):
// lanes 0-15: V0..3 -> K=0..7, V4..7 -> K=16..23 ; lanes 16-31: +8
static __device__ __forceinline__ int b_kmap(int v, int half) {
    return 2 * (v & 3) + ((v & 4) ? 16 : 0) + 8 * half;
}

// ---------------------------------------------------------------------------
// Elementwise fp32 -> bf16
// ---------------------------------------------------------------------------
__global__ __launch_bounds__(256) void f32_to_bf16_kernel(
    const float* __restrict__ in, bf16* __restrict__ out, int n)
{
    int i = blockIdx.x * blockDim.x + threadIdx.x;
    if (i < n) out[i] = (bf16)in[i];
}

// W [K,N] fp32 row-major  ->  Wt [N,K] bf16 row-major
__global__ __launch_bounds__(256) void transpose_to_bf16_kernel(
    const float* __restrict__ W, bf16* __restrict__ Wt, int N, int K)
{
    int t = blockIdx.x * blockDim.x + threadIdx.x;
    if (t < N * K) {
        int n = t / K, k = t - n * K;
        Wt[(size_t)n * K + k] = (bf16)W[(size_t)k * N + n];
    }
}

// ---------------------------------------------------------------------------
// C[M,N] = A[M,K](bf16) * Bt[N,K]^T(bf16) + bias  (fp32 out)
// Block = 8 waves = 128 rows x 64 cols. B-tile (64x32, 4KB) staged in LDS,
// DOUBLE-BUFFERED: one barrier per k-step, next tile's global load overlaps
// the current step's WMMAs. B frags preloaded into 4 independent register
// sets so their ds_loads pipeline instead of serializing on s_wait_dscnt.
// ---------------------------------------------------------------------------
__global__ __launch_bounds__(256) void gemm_bf16_wmma_kernel(
    const bf16* __restrict__ A, const bf16* __restrict__ Bt,
    const float* __restrict__ bias, float* __restrict__ C,
    int M, int N, int K)
{
    __shared__ __align__(16) bf16 lds_b[2][64][32];   // [buf][col][k] 8KB

    const int tid  = threadIdx.x;
    const int lane = tid & 31;
    const int wave = tid >> 5;
    const int half = lane >> 4;
    const int r    = lane & 15;

    const int rowBase = blockIdx.y * 128 + wave * 16;
    const int colBase = blockIdx.x * 64;

    // cooperative B-tile staging: 64 cols x 32 k, 8 bf16 (16B) per thread
    const int bcol = tid >> 2;        // 64 cols, 4 threads per col
    const int bk   = (tid & 3) * 8;   // 8 contiguous k elements
    const bf16* bsrc = &Bt[(size_t)(colBase + bcol) * K + bk];

    v8f acc[4];
#pragma unroll
    for (int t = 0; t < 4; ++t) acc[t] = (v8f)0.0f;

    bf16x8 breg = *reinterpret_cast<const bf16x8*>(bsrc);  // k0 = 0 tile
    int cur = 0;

    for (int k0 = 0; k0 < K; k0 += 32) {
        *reinterpret_cast<bf16x8*>(&lds_b[cur][bcol][bk]) = breg;
        __syncthreads();   // staged tile visible; prior reads of 'cur' done

        if (k0 + 32 < K)   // prefetch next tile, overlaps WMMA sequence
            breg = *reinterpret_cast<const bf16x8*>(bsrc + k0 + 32);

        v16bf afrag;
#pragma unroll
        for (int v = 0; v < 8; ++v) {
            int kk = a_kmap(v, half);
            bf16x2 p = *reinterpret_cast<const bf16x2*>(
                &A[(size_t)(rowBase + r) * K + k0 + kk]);
            afrag[2 * v]     = p.x;
            afrag[2 * v + 1] = p.y;
        }

        v16bf bfrag[4];
#pragma unroll
        for (int t = 0; t < 4; ++t) {
            int col = t * 16 + r;
#pragma unroll
            for (int v = 0; v < 8; ++v) {
                int kk = b_kmap(v, half);
                bf16x2 p = *reinterpret_cast<const bf16x2*>(&lds_b[cur][col][kk]);
                bfrag[t][2 * v]     = p.x;
                bfrag[t][2 * v + 1] = p.y;
            }
        }
#pragma unroll
        for (int t = 0; t < 4; ++t)
            acc[t] = __builtin_amdgcn_wmma_f32_16x16x32_bf16(
                false, afrag, false, bfrag[t], (short)0, acc[t], false, false);

        cur ^= 1;
    }

#pragma unroll
    for (int t = 0; t < 4; ++t) {
        int col = colBase + t * 16 + r;
        float bcol_f = bias[col];
#pragma unroll
        for (int vr = 0; vr < 8; ++vr) {
            int row = rowBase + vr + 8 * half;
            C[(size_t)row * N + col] = acc[t][vr] + bcol_f;
        }
    }
}

// ---------------------------------------------------------------------------
// RoPE on Q,K + repack q/k/v  [B,S,H*D] fp32 -> [B,H,S,D] bf16
// Q is pre-scaled by 1/sqrt(HD) = 0.125 (exact power of two) so the attention
// kernel needs no logit scaling. One thread per (b,s,h,i) pair, i in [0,32).
// ---------------------------------------------------------------------------
__global__ __launch_bounds__(256) void rope_pack_kernel(
    const float* __restrict__ Qf, const float* __restrict__ Kf,
    const float* __restrict__ Vf,
    bf16* __restrict__ qpk, bf16* __restrict__ kpk, bf16* __restrict__ vpk)
{
    int t = blockIdx.x * blockDim.x + threadIdx.x;
    if (t >= B_SZ * SEQ * NH * 32) return;
    int i = t & 31;
    int h = (t >> 5) & (NH - 1);
    int s = (t >> 9) & (SEQ - 1);
    int b = t >> 20;

    // inv_freq = 10000^(-2i/64)
    const float LOG_BASE = 9.2103403719761836f;  // ln(10000)
    float inv = __expf(-((float)(2 * i) / (float)HD) * LOG_BASE);
    float ang = (float)s * inv;
    float sn, cs;
    __sincosf(ang, &sn, &cs);

    size_t src = ((size_t)(b * SEQ + s)) * DM + h * HD;
    size_t dst = ((size_t)(b * NH + h) * SEQ + s) * HD;

    const float qscale = 0.125f;  // 1/sqrt(HEAD_DIM)
    float q1 = Qf[src + i], q2 = Qf[src + i + 32];
    qpk[dst + i]      = (bf16)((q1 * cs - q2 * sn) * qscale);
    qpk[dst + i + 32] = (bf16)((q1 * sn + q2 * cs) * qscale);

    float k1 = Kf[src + i], k2 = Kf[src + i + 32];
    kpk[dst + i]      = (bf16)(k1 * cs - k2 * sn);
    kpk[dst + i + 32] = (bf16)(k1 * sn + k2 * cs);

    vpk[dst + i]      = (bf16)Vf[src + i];
    vpk[dst + i + 32] = (bf16)Vf[src + i + 32];
}

// ---------------------------------------------------------------------------
// Flash attention. grid = (SEQ/128, B*NH), block = 256 (8 waves).
// Shared 32x64 K tile (row-major) and V tile (transposed [d][key]) are
// cooperatively staged in DOUBLE-BUFFERED LDS: one barrier per iteration,
// next tiles' global loads overlap the current WMMA/softmax work.
// Softmax state is per-wave: wave32 LDS ops are in-order (DScnt), so only
// a zero-cost wave_barrier (compiler fence) is needed there.
// ---------------------------------------------------------------------------
__global__ __launch_bounds__(256) void flash_attn_wmma_kernel(
    const bf16* __restrict__ qpk, const bf16* __restrict__ kpk,
    const bf16* __restrict__ vpk, float* __restrict__ out)
{
    __shared__ __align__(16) bf16  lds_k[2][32][64];      // [buf][key][d]  8KB
    __shared__ __align__(16) bf16  lds_vT[2][64][32];     // [buf][d][key]  8KB
    __shared__ __align__(16) float lds_logits[8][16][32]; // per-wave 16KB
    __shared__ __align__(16) bf16  lds_p[8][16][32];      // per-wave  8KB
    __shared__ float lds_alpha[8][16];
    __shared__ float lds_l[8][16];

    const int tid  = threadIdx.x;
    const int lane = tid & 31;
    const int wave = tid >> 5;
    const int half = lane >> 4;
    const int r    = lane & 15;

    const int bh = blockIdx.y;          // b*NH + h
    const int b  = bh >> 4;
    const int h  = bh & (NH - 1);
    const int qBase = blockIdx.x * 128 + wave * 16;

    const bf16* qhead = qpk + (size_t)bh * SEQ * HD;
    const bf16* khead = kpk + (size_t)bh * SEQ * HD;
    const bf16* vhead = vpk + (size_t)bh * SEQ * HD;

    // cooperative staging: 32 keys x 64 d, 8 bf16 (16B) per thread
    const int krow = tid >> 3;        // 32 key rows, 8 threads per row
    const int kcol = (tid & 7) * 8;   // 8 contiguous d elements
    const bf16* ksrc = &khead[(size_t)krow * HD + kcol];
    const bf16* vsrc = &vhead[(size_t)krow * HD + kcol];

    // Q fragments (pre-scaled by 1/sqrt(HD)): 2 d-chunks of 32
    v16bf qf[2];
#pragma unroll
    for (int dc = 0; dc < 2; ++dc) {
#pragma unroll
        for (int v = 0; v < 8; ++v) {
            int kk = a_kmap(v, half);
            bf16x2 p = *reinterpret_cast<const bf16x2*>(
                &qhead[(size_t)(qBase + r) * HD + dc * 32 + kk]);
            qf[dc][2 * v]     = p.x;
            qf[dc][2 * v + 1] = p.y;
        }
    }

    v8f acc[4];
#pragma unroll
    for (int t = 0; t < 4; ++t) acc[t] = (v8f)0.0f;
    float m_run = -3.0e38f, l_run = 0.0f;

    bf16x8 kreg = *reinterpret_cast<const bf16x8*>(ksrc);  // key0 = 0 tiles
    bf16x8 vreg = *reinterpret_cast<const bf16x8*>(vsrc);
    int cur = 0;

    for (int key0 = 0; key0 < SEQ; key0 += 32) {
        // ---- commit staged tiles, single barrier ----
        *reinterpret_cast<bf16x8*>(&lds_k[cur][krow][kcol]) = kreg;
#pragma unroll
        for (int j = 0; j < 8; ++j)
            lds_vT[cur][kcol + j][krow] = vreg[j];
        __syncthreads();   // tiles visible; prior reads of 'cur' done

        if (key0 + 32 < SEQ) {   // prefetch next tiles, overlaps compute
            kreg = *reinterpret_cast<const bf16x8*>(ksrc + (key0 + 32) * HD);
            vreg = *reinterpret_cast<const bf16x8*>(vsrc + (key0 + 32) * HD);
        }

        // ---- logits tile 16x32 = 2 key groups x 2 d-chunks of WMMA ----
#pragma unroll
        for (int g = 0; g < 2; ++g) {
            v16bf kf[2];
#pragma unroll
            for (int dc = 0; dc < 2; ++dc) {
                int key = g * 16 + r;
#pragma unroll
                for (int v = 0; v < 8; ++v) {
                    int kk = b_kmap(v, half);
                    bf16x2 p = *reinterpret_cast<const bf16x2*>(
                        &lds_k[cur][key][dc * 32 + kk]);
                    kf[dc][2 * v]     = p.x;
                    kf[dc][2 * v + 1] = p.y;
                }
            }
            v8f c = (v8f)0.0f;
            c = __builtin_amdgcn_wmma_f32_16x16x32_bf16(
                false, qf[0], false, kf[0], (short)0, c, false, false);
            c = __builtin_amdgcn_wmma_f32_16x16x32_bf16(
                false, qf[1], false, kf[1], (short)0, c, false, false);
#pragma unroll
            for (int vr = 0; vr < 8; ++vr)
                lds_logits[wave][vr + 8 * half][g * 16 + r] = c[vr];
        }
        __builtin_amdgcn_wave_barrier();  // per-wave LDS: in-order, fence only

        // ---- online softmax: lane (r,half) handles row r, cols half*16.. ----
        {
            int row   = r;
            int jbase = half * 16;
            float mx = -3.0e38f;
#pragma unroll
            for (int j = 0; j < 16; ++j)
                mx = fmaxf(mx, lds_logits[wave][row][jbase + j]);
            mx = fmaxf(mx, __shfl_xor(mx, 16, 32));      // combine halves
            float m_new = fmaxf(m_run, mx);
            float alpha = __expf(m_run - m_new);
            float s_loc = 0.0f;
#pragma unroll
            for (int j = 0; j < 16; ++j) {
                float pv = __expf(lds_logits[wave][row][jbase + j] - m_new);
                s_loc += pv;
                lds_p[wave][row][jbase + j] = (bf16)pv;
            }
            float s = s_loc + __shfl_xor(s_loc, 16, 32); // combine halves
            l_run = l_run * alpha + s;
            m_run = m_new;
            lds_alpha[wave][row] = alpha;  // both halves write same value
        }
        __builtin_amdgcn_wave_barrier();

        // ---- rescale O accumulator ----
#pragma unroll
        for (int t = 0; t < 4; ++t)
#pragma unroll
            for (int vr = 0; vr < 8; ++vr)
                acc[t][vr] *= lds_alpha[wave][vr + 8 * half];

        // ---- P (A-frag via per-wave LDS transpose) x V ----
        v16bf pf;
#pragma unroll
        for (int v = 0; v < 8; ++v) {
            int kk = a_kmap(v, half);
            bf16x2 p = *reinterpret_cast<const bf16x2*>(&lds_p[wave][r][kk]);
            pf[2 * v]     = p.x;
            pf[2 * v + 1] = p.y;
        }
        v16bf vf[4];
#pragma unroll
        for (int t = 0; t < 4; ++t) {
            int d = t * 16 + r;
#pragma unroll
            for (int v = 0; v < 8; ++v) {
                int kk = b_kmap(v, half);
                bf16x2 p = *reinterpret_cast<const bf16x2*>(&lds_vT[cur][d][kk]);
                vf[t][2 * v]     = p.x;
                vf[t][2 * v + 1] = p.y;
            }
        }
#pragma unroll
        for (int t = 0; t < 4; ++t)
            acc[t] = __builtin_amdgcn_wmma_f32_16x16x32_bf16(
                false, pf, false, vf[t], (short)0, acc[t], false, false);

        cur ^= 1;
    }

    lds_l[wave][r] = l_run;   // both halves write same value
    __builtin_amdgcn_wave_barrier();

#pragma unroll
    for (int t = 0; t < 4; ++t) {
#pragma unroll
        for (int vr = 0; vr < 8; ++vr) {
            int row = vr + 8 * half;
            int q   = qBase + row;
            int d   = t * 16 + r;
            float val = acc[t][vr] / lds_l[wave][row];
            out[((size_t)(b * SEQ + q)) * DM + h * HD + d] = val;
        }
    }
}

// ---------------------------------------------------------------------------
// Host orchestration
// ---------------------------------------------------------------------------
extern "C" void kernel_launch(void* const* d_in, const int* in_sizes, int n_in,
                              void* d_out, int out_size, void* d_ws, size_t ws_size,
                              hipStream_t stream) {
    const float* x  = (const float*)d_in[0];
    const float* Wq = (const float*)d_in[1];
    const float* bq = (const float*)d_in[2];
    const float* Wk = (const float*)d_in[3];
    const float* bk = (const float*)d_in[4];
    const float* Wv = (const float*)d_in[5];
    const float* bv = (const float*)d_in[6];
    const float* Wo = (const float*)d_in[7];
    const float* bo = (const float*)d_in[8];
    float* out = (float*)d_out;

    char* ws = (char*)d_ws;
    size_t off = 0;
    auto take = [&](size_t bytes) -> char* {
        char* p = ws + off;
        off = (off + bytes + 255) & ~(size_t)255;
        return p;
    };

    const size_t MD = (size_t)M_ROWS * DM;     // 4096*1024 elements
    const size_t WW = (size_t)DM * DM;

    bf16*  xb    = (bf16*)take(MD * sizeof(bf16));
    bf16*  WqT   = (bf16*)take(WW * sizeof(bf16));
    bf16*  WkT   = (bf16*)take(WW * sizeof(bf16));
    bf16*  WvT   = (bf16*)take(WW * sizeof(bf16));
    bf16*  WoT   = (bf16*)take(WW * sizeof(bf16));
    float* Qf    = (float*)take(MD * sizeof(float));
    float* Kf    = (float*)take(MD * sizeof(float));
    float* Vf    = (float*)take(MD * sizeof(float));
    bf16*  qpk   = (bf16*)take(MD * sizeof(bf16));
    bf16*  kpk   = (bf16*)take(MD * sizeof(bf16));
    bf16*  vpk   = (bf16*)take(MD * sizeof(bf16));
    float* attnf = (float*)take(MD * sizeof(float));
    bf16*  attnb = (bf16*)take(MD * sizeof(bf16));

    // 1. convert x, transpose-convert weights
    f32_to_bf16_kernel<<<(int)(MD / 256), 256, 0, stream>>>(x, xb, (int)MD);
    int wgridT = (int)((WW + 255) / 256);
    transpose_to_bf16_kernel<<<wgridT, 256, 0, stream>>>(Wq, WqT, DM, DM);
    transpose_to_bf16_kernel<<<wgridT, 256, 0, stream>>>(Wk, WkT, DM, DM);
    transpose_to_bf16_kernel<<<wgridT, 256, 0, stream>>>(Wv, WvT, DM, DM);
    transpose_to_bf16_kernel<<<wgridT, 256, 0, stream>>>(Wo, WoT, DM, DM);

    // 2. Q/K/V projections (WMMA bf16, fp32 accumulate + bias)
    dim3 ggrid(DM / 64, M_ROWS / 128);
    gemm_bf16_wmma_kernel<<<ggrid, 256, 0, stream>>>(xb, WqT, bq, Qf, M_ROWS, DM, DM);
    gemm_bf16_wmma_kernel<<<ggrid, 256, 0, stream>>>(xb, WkT, bk, Kf, M_ROWS, DM, DM);
    gemm_bf16_wmma_kernel<<<ggrid, 256, 0, stream>>>(xb, WvT, bv, Vf, M_ROWS, DM, DM);

    // 3. RoPE + repack to [B,H,S,D] bf16 (Q pre-scaled by 1/sqrt(HD))
    int ropeN = B_SZ * SEQ * NH * 32;
    rope_pack_kernel<<<ropeN / 256, 256, 0, stream>>>(Qf, Kf, Vf, qpk, kpk, vpk);

    // 4. flash attention (WMMA bf16, fp32 softmax)
    dim3 agrid(SEQ / 128, B_SZ * NH);
    flash_attn_wmma_kernel<<<agrid, 256, 0, stream>>>(qpk, kpk, vpk, attnf);

    // 5. output projection
    f32_to_bf16_kernel<<<(int)(MD / 256), 256, 0, stream>>>(attnf, attnb, (int)MD);
    gemm_bf16_wmma_kernel<<<ggrid, 256, 0, stream>>>(attnb, WoT, bo, out, M_ROWS, DM, DM);
}
